// GNN_78589311582920
// MI455X (gfx1250) — matrix operand order repaired and checked by reading, not compiled, not run on hardware
//
#include <hip/hip_runtime.h>
#include <hip/hip_bf16.h>

// ---------------------------------------------------------------------------
// MI455X (gfx1250) implementation.
// Compute core: v_wmma_f32_16x16x32_bf16 (wave32 WMMA, f32 accumulate).
// GEMM tiles are staged into LDS by the Tensor Data Mover
// (tensor_load_to_lds, TENSORcnt) with double buffering: DMA of tile t+1
// overlaps WMMA compute of tile t. Weights pre-transposed to bf16 once.
// Scatter phase (bandwidth-bound) uses coalesced per-edge-channel atomics.
// ---------------------------------------------------------------------------

typedef __bf16 bf16_t;
typedef bf16_t bf16x16 __attribute__((ext_vector_type(16)));
typedef float  floatx8 __attribute__((ext_vector_type(8)));
typedef unsigned int uint32x4 __attribute__((ext_vector_type(4)));
typedef int int32x8 __attribute__((ext_vector_type(8)));
typedef int int32x4 __attribute__((ext_vector_type(4)));

__device__ __forceinline__ unsigned short f2bf(float f) {
    unsigned int u = __float_as_uint(f);
    unsigned int r = u + 0x7FFFu + ((u >> 16) & 1u);   // round-to-nearest-even
    return (unsigned short)(r >> 16);
}

struct U4x2 { uint4 a, b; };

__device__ __forceinline__ bf16x16 ldfrag(const unsigned short* p0,
                                          const unsigned short* p1) {
    U4x2 t;
    t.a = *(const uint4*)p0;
    t.b = *(const uint4*)p1;
    return __builtin_bit_cast(bf16x16, t);
}

// Generic pointers to __shared__ carry the LDS offset in bits [31:0]
// (ISA 10.2 aperture mapping), which is what the TDM D# lds_addr wants.
__device__ __forceinline__ unsigned lds_lo(const void* p) {
    return (unsigned)(unsigned long long)p;
}

// ---------------------------------------------------------------------------
// TDM: async 2D tile load (rows x 64 bf16) from a row-major matrix into LDS.
//   D# group0: count=1 | lds_addr | global_addr(57b) | type=2
//   D# group1: mask=0, data_size=1(2B), pad_enable, pad_interval=4 (32 dwords
//              = one 128B row), pad_amount=3 (4 dwords = 16B) -> LDS row
//              stride 144B; tensor_dim0=K, tensor_dim1=rows, tile=64x64,
//              dim0_stride=K.
//   Groups 2/3 + trailing group zero (2D tensor). Tracked by TENSORcnt.
//   This toolchain exposes the 6-operand builtin:
//     (uint32x4, int32x8, int32x4, int32x4, int32x8, i32 cpol)
// ---------------------------------------------------------------------------
__device__ __forceinline__ void tdm_load_tile(const unsigned short* gbase,
                                              unsigned ldsOff, int K, int rows,
                                              int row0, int k0) {
    unsigned long long ga = (unsigned long long)gbase +
        (((unsigned long long)row0 * (unsigned)K + (unsigned)k0) << 1);
    uint32x4 g0;
    g0[0] = 1u;                                        // count=1
    g0[1] = ldsOff;                                    // lds_addr
    g0[2] = (unsigned)(ga & 0xFFFFFFFFull);            // global_addr[31:0]
    g0[3] = (unsigned)((ga >> 32) & 0x1FFFFFFull) | (2u << 30);  // [56:32]|type
    int32x8 g1;
    g1[0] = (int)((1u << 16) | (1u << 20) | (4u << 22) | (3u << 25));
    g1[1] = (int)(((unsigned)K & 0xFFFFu) << 16);      // tensor_dim0 lo16
    g1[2] = (int)((((unsigned)K >> 16) & 0xFFFFu) |
                  (((unsigned)rows & 0xFFFFu) << 16)); // dim0 hi | dim1 lo
    g1[3] = (int)((((unsigned)rows >> 16) & 0xFFFFu) | (64u << 16)); // |tile0
    g1[4] = 64;                                        // tile_dim1 (tile2=0)
    g1[5] = K;                                         // dim0_stride[31:0]
    g1[6] = 0;
    g1[7] = 0;
    int32x4 z4 = {0, 0, 0, 0};
    int32x8 z8 = {0, 0, 0, 0, 0, 0, 0, 0};
    __builtin_amdgcn_tensor_load_to_lds(g0, g1, z4, z4, z8, 0);
}

// ---------------------------------------------------------------------------
// Weight convert + transpose: W (K x N, f32, row-major) -> WT (N x K, bf16)
// ---------------------------------------------------------------------------
__global__ void wconv_kernel(const float* __restrict__ W,
                             unsigned short* __restrict__ WT, int K, int N) {
    int idx = blockIdx.x * blockDim.x + threadIdx.x;
    if (idx >= K * N) return;
    int k = idx / N, n = idx % N;
    WT[(size_t)n * K + k] = f2bf(W[idx]);
}

// flat f32 -> bf16
__global__ void aconv_kernel(const float* __restrict__ X,
                             unsigned short* __restrict__ Y, int n) {
    int i = blockIdx.x * blockDim.x + threadIdx.x;
    if (i < n) Y[i] = f2bf(X[i]);
}

// ---------------------------------------------------------------------------
// Tiled WMMA GEMM: C(MxN) = act(A(MxK,bf16) @ W(KxN) + bias)
//   BM=BN=64, BK=64 (2 WMMA K-steps per tile), TDM-staged double-buffered
//   LDS. 256 threads = 8 wave32; wave w owns C subtiles (mi=w&3,
//   ni=2*(w>>2), +1). Fused epilogue:
//     Cb (optional): bf16 MxN output (feeds next layer)
//     Cf (optional): f32 output with node-row remap
//                    orow = (m/rps)*336 + base + (m%rps)
// ---------------------------------------------------------------------------
__global__ __launch_bounds__(256) void gemm_bf16_kernel(
    const unsigned short* __restrict__ A, const unsigned short* __restrict__ WT,
    const float* __restrict__ bias, unsigned short* __restrict__ Cb,
    float* __restrict__ Cf, int M, int N, int K, int relu, int base, int rps) {
    // 64 rows x (64 bf16 + 8 pad) per buffer; TDM pads 16B per 128B row.
    __shared__ unsigned short sA[2][64 * 72];
    __shared__ unsigned short sB[2][64 * 72];

    const int tid  = threadIdx.x;
    const int m0   = blockIdx.y * 64;
    const int n0   = blockIdx.x * 64;
    const int wv   = tid >> 5;
    const int lane = tid & 31;
    const int mi   = wv & 3;
    const int nA   = (wv >> 2) << 1;         // 0 or 2

    floatx8 acc0 = {};
    floatx8 acc1 = {};

    // WMMA fragment offsets (contiguous k = 16*khalf + s in both A and B, wave32), row stride 72
    const int frow  = lane & 15;
    const int khalf = lane >> 4;
    const unsigned aoff  = (unsigned)((mi * 16 + frow) * 72 + khalf * 16);
    const unsigned boff0 = (unsigned)(((nA)     * 16 + frow) * 72 + khalf * 16);
    const unsigned boff1 = (unsigned)(((nA + 1) * 16 + frow) * 72 + khalf * 16);

    const int nk = K >> 6;
    if (wv == 0) {
        tdm_load_tile(A,  lds_lo(&sA[0][0]), K, M, m0, 0);
        tdm_load_tile(WT, lds_lo(&sB[0][0]), K, N, n0, 0);
        __builtin_amdgcn_s_wait_tensorcnt(0);
    }
    __syncthreads();

    for (int t = 0; t < nk; ++t) {
        const int buf = t & 1;
        if ((t + 1 < nk) && (wv == 0)) {     // overlap next DMA with compute
            tdm_load_tile(A,  lds_lo(&sA[buf ^ 1][0]), K, M, m0, (t + 1) << 6);
            tdm_load_tile(WT, lds_lo(&sB[buf ^ 1][0]), K, N, n0, (t + 1) << 6);
        }
        const unsigned short* pa = &sA[buf][0];
        const unsigned short* pb = &sB[buf][0];
#pragma unroll
        for (int kk = 0; kk < 64; kk += 32) {
            bf16x16 af = ldfrag(pa + aoff + kk,  pa + aoff + kk + 8);
            bf16x16 b0 = ldfrag(pb + boff0 + kk, pb + boff0 + kk + 8);
            bf16x16 b1 = ldfrag(pb + boff1 + kk, pb + boff1 + kk + 8);
            acc0 = __builtin_amdgcn_wmma_f32_16x16x32_bf16(
                false, af, false, b0, (short)0, acc0, false, false);
            acc1 = __builtin_amdgcn_wmma_f32_16x16x32_bf16(
                false, af, false, b1, (short)0, acc1, false, false);
        }
        if (t + 1 < nk) {
            if (wv == 0) __builtin_amdgcn_s_wait_tensorcnt(0);
            __syncthreads();
        }
    }

    // C layout: lanes 0-15 hold M=j (VGPR j), lanes 16-31 hold M=8+j; N=lane&15
    const int cn0   = n0 + nA * 16 + frow;
    const int cn1   = cn0 + 16;
    const int rbase = m0 + mi * 16 + khalf * 8;
#pragma unroll
    for (int j = 0; j < 8; ++j) {
        int m = rbase + j;
        float v0 = acc0[j];
        float v1 = acc1[j];
        if (bias) { v0 += bias[cn0]; v1 += bias[cn1]; }
        if (relu) { v0 = fmaxf(v0, 0.f); v1 = fmaxf(v1, 0.f); }
        if (Cb) {
            Cb[(size_t)m * N + cn0] = f2bf(v0);
            Cb[(size_t)m * N + cn1] = f2bf(v1);
        }
        if (Cf) {
            int orow = (m / rps) * 336 + base + (m % rps);
            Cf[(size_t)orow * N + cn0] = v0;
            Cf[(size_t)orow * N + cn1] = v1;
        }
    }
}

// ---------------------------------------------------------------------------
// Sinusoidal block pos-encoding + learned pos_embed, added to both modalities
// ---------------------------------------------------------------------------
__global__ void add_pos_kernel(float* __restrict__ xa, float* __restrict__ xp,
                               const float* __restrict__ pe, int total) {
    int idx = blockIdx.x * blockDim.x + threadIdx.x;
    if (idx >= total) return;
    int c = idx & 63, node = idx >> 6;
    int w = node % 336, b = node / 336;
    float p = 0.f;
    if (w >= 16) {
        int k = (w >= 80) ? (w - 80) >> 4 : (w - 16) >> 4;
        int i = c >> 1;
        float ang = (float)k * __powf(10000.0f, -(float)(2 * i) / 64.0f);
        p = (c & 1) ? __cosf(ang) : __sinf(ang);
    }
    float v = p + pe[(size_t)b * 336 * 64 + w * 64 + c];
    xa[idx] += v;
    xp[idx] += v;
}

// ---------------------------------------------------------------------------
// GCN normalization + aggregation
// ---------------------------------------------------------------------------
__global__ void deg_init_kernel(float* deg, int n) {
    int i = blockIdx.x * blockDim.x + threadIdx.x;
    if (i < n) deg[i] = 1.0f;                 // self loop
}
__global__ void deg_count_kernel(const int* __restrict__ dst, float* deg, int E) {
    int i = blockIdx.x * blockDim.x + threadIdx.x;
    if (i < E) atomicAdd(&deg[dst[i]], 1.0f);
}
__global__ void dinv_kernel(float* deg, int n) {
    int i = blockIdx.x * blockDim.x + threadIdx.x;
    if (i < n) deg[i] = rsqrtf(deg[i]);
}
__global__ void agg_init_kernel(const float* __restrict__ h,
                                const float* __restrict__ dinv,
                                float* __restrict__ out, int total) {
    int idx = blockIdx.x * blockDim.x + threadIdx.x;
    if (idx >= total) return;
    float d = dinv[idx >> 6];
    out[idx] = d * d * h[idx];                // self-loop contribution
}
__global__ void agg_edge_kernel(const int* __restrict__ src,
                                const int* __restrict__ dst,
                                const float* __restrict__ h,
                                const float* __restrict__ dinv,
                                float* __restrict__ out, int E) {
    int idx = blockIdx.x * blockDim.x + threadIdx.x;
    if (idx >= E * 64) return;
    int e = idx >> 6, c = idx & 63;
    int s = src[e], d = dst[e];
    atomicAdd(&out[(size_t)d * 64 + c], dinv[s] * dinv[d] * h[(size_t)s * 64 + c]);
}

// relu(agg + bias) then mean over the 336 nodes of each sample
__global__ void pool_kernel(const float* __restrict__ agg,
                            const float* __restrict__ bias,
                            float* __restrict__ pooled, int Bv) {
    int idx = blockIdx.x * blockDim.x + threadIdx.x;
    if (idx >= Bv * 64) return;
    int s = idx >> 6, c = idx & 63;
    const float* p = agg + (size_t)s * 336 * 64 + c;
    float b = bias[c];
    float acc = 0.f;
    for (int j = 0; j < 336; ++j) acc += fmaxf(p[j * 64] + b, 0.f);
    pooled[idx] = acc * (1.0f / 336.0f);
}

// conv1d(2->1,k=1) + linear(64->10) + softmax
__global__ void head_kernel(const float* __restrict__ pa, const float* __restrict__ pp,
                            const float* __restrict__ cw, const float* __restrict__ cb,
                            const float* __restrict__ lw, const float* __restrict__ lb,
                            float* __restrict__ out) {
    int s = blockIdx.x, t = threadIdx.x;
    __shared__ float x[64];
    __shared__ float lg[10];
    x[t] = cw[0] * pa[s * 64 + t] + cw[1] * pp[s * 64 + t] + cb[0];
    __syncthreads();
    if (t < 10) {
        float acc = lb[t];
        for (int c = 0; c < 64; ++c) acc += x[c] * lw[c * 10 + t];
        lg[t] = acc;
    }
    __syncthreads();
    if (t < 10) {
        float mx = lg[0];
        for (int i = 1; i < 10; ++i) mx = fmaxf(mx, lg[i]);
        float sm = 0.f;
        for (int i = 0; i < 10; ++i) sm += __expf(lg[i] - mx);
        out[s * 10 + t] = __expf(lg[t] - mx) / sm;
    }
}

// ---------------------------------------------------------------------------
// Host orchestration
// ---------------------------------------------------------------------------
extern "C" void kernel_launch(void* const* d_in, const int* in_sizes, int n_in,
                              void* d_out, int out_size, void* d_ws, size_t ws_size,
                              hipStream_t stream) {
    (void)n_in; (void)out_size; (void)ws_size;
    static const int d0[] = {4096, 2048, 1024, 512, 256, 128, 64};
    static const int d1[] = {1024, 512, 256, 128, 64};
    static const int d2[] = {256, 128, 64};
    const int* dims[3]  = {d0, d1, d2};
    const int  nl[3]    = {6, 4, 2};
    const int  chOff[3] = {0, 12, 20};   // leaf offset inside one enc_* list
    const int  Mch[3]   = {960, 3840, 15360};
    const int  baseCh[3] = {0, 16, 80};
    const int  rpsCh[3]  = {16, 64, 256};

    char* ws = (char*)d_ws;
    size_t off = 0;
    auto alloc = [&](size_t bytes) -> char* {
        size_t o = (off + 255) & ~(size_t)255;
        off = o + bytes;
        return ws + o;
    };

    unsigned short* wt[2][3][6];
    for (int md = 0; md < 2; ++md)
        for (int c = 0; c < 3; ++c)
            for (int l = 0; l < nl[c]; ++l)
                wt[md][c][l] =
                    (unsigned short*)alloc((size_t)dims[c][l] * dims[c][l + 1] * 2);
    unsigned short* gwt[2];
    gwt[0] = (unsigned short*)alloc(64 * 64 * 2);
    gwt[1] = (unsigned short*)alloc(64 * 64 * 2);

    unsigned short* actA = (unsigned short*)alloc((size_t)3932160 * 2);
    unsigned short* actB = (unsigned short*)alloc((size_t)3932160 * 2);

    const int Nn = in_sizes[64];          // 20160 nodes
    const int Bv = Nn / 336;              // 60 samples
    const int E  = in_sizes[63] / 2;      // 645120 edges
    const int totN = Nn * 64;

    float* X[2];
    X[0] = (float*)alloc((size_t)totN * 4);
    X[1] = (float*)alloc((size_t)totN * 4);
    float* h    = (float*)alloc((size_t)totN * 4);
    float* agg  = (float*)alloc((size_t)totN * 4);
    float* dinv = (float*)alloc((size_t)Nn * 4);
    float* pooled[2];
    pooled[0] = (float*)alloc((size_t)Bv * 64 * 4);
    pooled[1] = (float*)alloc((size_t)Bv * 64 * 4);

    // --- 1. weights -> transposed bf16 ---
    for (int md = 0; md < 2; ++md)
        for (int c = 0; c < 3; ++c)
            for (int l = 0; l < nl[c]; ++l) {
                int K = dims[c][l], Nw = dims[c][l + 1];
                const float* W = (const float*)d_in[6 + md * 24 + chOff[c] + 2 * l];
                int tot = K * Nw;
                wconv_kernel<<<(tot + 255) / 256, 256, 0, stream>>>(W, wt[md][c][l], K, Nw);
            }
    wconv_kernel<<<(64 * 64 + 255) / 256, 256, 0, stream>>>(
        (const float*)d_in[55], gwt[0], 64, 64);
    wconv_kernel<<<(64 * 64 + 255) / 256, 256, 0, stream>>>(
        (const float*)d_in[57], gwt[1], 64, 64);

    // --- 2. encoder MLP chains (WMMA GEMMs, fused bias/ReLU, final-layer remap) ---
    for (int md = 0; md < 2; ++md) {
        for (int c = 0; c < 3; ++c) {
            const float* inp = (const float*)d_in[md * 3 + c];
            int M = Mch[c];
            int tot = M * dims[c][0];
            aconv_kernel<<<(tot + 255) / 256, 256, 0, stream>>>(inp, actA, tot);
            unsigned short* cur = actA;
            unsigned short* nxt = actB;
            for (int l = 0; l < nl[c]; ++l) {
                int K = dims[c][l], Nw = dims[c][l + 1];
                const float* bias =
                    (const float*)d_in[6 + md * 24 + chOff[c] + 2 * l + 1];
                bool last = (l == nl[c] - 1);
                dim3 grid(Nw / 64, M / 64);
                gemm_bf16_kernel<<<grid, 256, 0, stream>>>(
                    cur, wt[md][c][l], bias,
                    last ? nullptr : nxt,
                    last ? X[md] : nullptr,
                    M, Nw, K, /*relu=*/1, baseCh[c], rpsCh[c]);
                unsigned short* t = cur; cur = nxt; nxt = t;
            }
        }
    }

    // --- 3. positional encodings ---
    add_pos_kernel<<<(totN + 255) / 256, 256, 0, stream>>>(
        X[0], X[1], (const float*)d_in[54], totN);

    // --- 4. degrees / dinv ---
    const int* ei = (const int*)d_in[63];
    deg_init_kernel<<<(Nn + 255) / 256, 256, 0, stream>>>(dinv, Nn);
    deg_count_kernel<<<(E + 255) / 256, 256, 0, stream>>>(ei + E, dinv, E);
    dinv_kernel<<<(Nn + 255) / 256, 256, 0, stream>>>(dinv, Nn);

    // --- 5. GCN per modality: X@W (WMMA) -> normalized scatter -> pooled ---
    for (int md = 0; md < 2; ++md) {
        aconv_kernel<<<(totN + 255) / 256, 256, 0, stream>>>(X[md], actA, totN);
        dim3 g(1, Nn / 64);
        gemm_bf16_kernel<<<g, 256, 0, stream>>>(
            actA, gwt[md], nullptr, nullptr, h,
            Nn, 64, 64, /*relu=*/0, /*base=*/0, /*rps=*/336);  // identity remap
        agg_init_kernel<<<(totN + 255) / 256, 256, 0, stream>>>(h, dinv, agg, totN);
        long long te = (long long)E * 64;
        agg_edge_kernel<<<(int)((te + 255) / 256), 256, 0, stream>>>(
            ei, ei + E, h, dinv, agg, E);
        pool_kernel<<<(Bv * 64 + 255) / 256, 256, 0, stream>>>(
            agg, (const float*)d_in[md == 0 ? 56 : 58], pooled[md], Bv);
    }

    // --- 6. head ---
    head_kernel<<<Bv, 64, 0, stream>>>(
        pooled[0], pooled[1], (const float*)d_in[59], (const float*)d_in[60],
        (const float*)d_in[61], (const float*)d_in[62], (float*)d_out);
}
